// AttnModule_21629455303041
// MI455X (gfx1250) — matrix operand order repaired and checked by reading, not compiled, not run on hardware
//
#include <hip/hip_runtime.h>
#include <math.h>

// Problem dims (fixed by the reference)
#define B_  64
#define L_  1024
#define H_  512
#define T_  100
#define C_  97

typedef __attribute__((ext_vector_type(16))) __bf16        v16bf;
typedef __attribute__((ext_vector_type(8)))  float         v8f;
typedef __attribute__((ext_vector_type(4)))  unsigned int  v4u;
typedef __attribute__((ext_vector_type(8)))  int           v8i;
typedef __attribute__((ext_vector_type(4)))  int           v4i;

union Frag16 {
  v16bf          v;
  unsigned short u[16];
  uint4          q[2];
};

// round-to-nearest-even f32 -> bf16
__device__ __forceinline__ unsigned short f2bf(float f) {
  unsigned int u = __builtin_bit_cast(unsigned int, f);
  u += 0x7FFFu + ((u >> 16) & 1u);
  return (unsigned short)(u >> 16);
}

__device__ __forceinline__ void cvt4(unsigned short* d, float4 f) {
  d[0] = f2bf(f.x); d[1] = f2bf(f.y); d[2] = f2bf(f.z); d[3] = f2bf(f.w);
}

// D = A(16x32 bf16) * B(32x16 bf16) + C(f32)
__device__ __forceinline__ v8f wmma_bf16(const Frag16& a, const Frag16& b, v8f c) {
  return __builtin_amdgcn_wmma_f32_16x16x32_bf16(
      /*neg_a=*/false, a.v, /*neg_b=*/false, b.v,
      /*c_mod=*/(short)0, c, /*reuse_a=*/false, /*reuse_b=*/false);
}

__global__ __launch_bounds__(256)
void attn_gen_kernel(const float* __restrict__ Kg,   // position_fmap [B,L,H]
                     const float* __restrict__ Vg,   // origin_fmap   [B,L,H]
                     const float* __restrict__ Qg,   // pos_emb       [T,H]
                     const float* __restrict__ Wg,   // W_gen         [C,H]
                     const float* __restrict__ bg,   // b_gen         [C]
                     float* __restrict__ out)        // out           [B,T,C]
{
  // LDS: 64 + 16 + 32 + 16 = 128 KB (within 320 KB/WG on CDNA5).
  // Ss (scores) is dead after softmax and is reused as the TDM V-staging tile.
  __shared__ float          Ss[16 * L_];   // scores f32 [16][1024] / Vstage f32 [32][512]
  __shared__ unsigned short Qs[16 * H_];   // Q tile bf16 [16][512]
  __shared__ unsigned short As[16 * L_];   // attn  bf16 [16][1024]
  __shared__ unsigned short Cs[16 * H_];   // ctx   bf16 [16][512]

  const int tid  = threadIdx.x;
  const int wave = tid >> 5;     // 8 wave32s
  const int lane = tid & 31;
  const int half = lane >> 4;    // 0: lanes 0-15, 1: lanes 16-31
  const int l16  = lane & 15;
  const int b    = blockIdx.y;
  const int t0   = blockIdx.x * 16;

  const float* Kb = Kg + (size_t)b * L_ * H_;
  const float* Vb = Vg + (size_t)b * L_ * H_;

  // ---- Phase 0: stage Q tile (16 x H) into LDS as bf16; pad t>=T with 0 ----
  for (int idx = tid; idx < 16 * H_; idx += 256) {
    const int m = idx >> 9;          // /512
    const int h = idx & (H_ - 1);
    const int t = t0 + m;
    Qs[idx] = f2bf((t < T_) ? Qg[t * H_ + h] : 0.0f);
  }
  __syncthreads();

  // ---- Phase A: S[16][1024] = Qtile * K^T, WMMA bf16 over H in 32-chunks ----
  for (int pass = 0; pass < 8; ++pass) {
    const int l0   = pass * 128 + wave * 16;
    const int lcol = l0 + l16;
    const unsigned short* qrow = &Qs[l16 << 9];   // A row = t-row (lane&15)
    v8f acc0 = {};
    v8f acc1 = {};
    for (int h0 = 0; h0 < H_; h0 += 64) {
#pragma unroll
      for (int s = 0; s < 2; ++s) {
        const int hh = h0 + s * 32;
        Frag16 a, bm;
        // A layout: lane m -> K hh + half*16 + j (contiguous in h)
        a.q[0] = *(const uint4*)(qrow + hh + half * 16);
        a.q[1] = *(const uint4*)(qrow + hh + half * 16 + 8);
        // B layout: lane n -> col lcol, K = hh + half*16 + j (contiguous in h)
        const float* kp = Kb + (size_t)lcol * H_ + hh + half * 16;
        __builtin_prefetch(kp + 256, 0, 0);        // global_prefetch_b8
        cvt4(&bm.u[0],  ((const float4*)kp)[0]);
        cvt4(&bm.u[4],  ((const float4*)kp)[1]);
        cvt4(&bm.u[8],  ((const float4*)kp)[2]);
        cvt4(&bm.u[12], ((const float4*)kp)[3]);
        if (s == 0) acc0 = wmma_bf16(a, bm, acc0);
        else        acc1 = wmma_bf16(a, bm, acc1);
      }
    }
    acc0 += acc1;
    // D layout: VGPR i, lane half -> row i + 8*half, col = lcol
#pragma unroll
    for (int i = 0; i < 8; ++i) {
      Ss[(i + 8 * half) * L_ + lcol] = acc0[i];
    }
  }
  __syncthreads();

  // ---- Phase B: softmax over L for each of 16 rows (2 rows per wave) ----
  for (int r = wave * 2; r < wave * 2 + 2; ++r) {
    const float* srow = &Ss[r * L_];
    float mx = -3.402823466e38f;
    for (int k = lane; k < L_; k += 32) mx = fmaxf(mx, srow[k]);
#pragma unroll
    for (int o = 16; o > 0; o >>= 1) mx = fmaxf(mx, __shfl_xor(mx, o, 32));
    float sum = 0.0f;
    for (int k = lane; k < L_; k += 32) sum += __expf(srow[k] - mx);
#pragma unroll
    for (int o = 16; o > 0; o >>= 1) sum += __shfl_xor(sum, o, 32);
    const float inv = 1.0f / sum;
    for (int k = lane; k < L_; k += 32)
      As[r * L_ + k] = f2bf(__expf(srow[k] - mx) * inv);
  }
  __syncthreads();

  // ---- Phase C: ctx[16][512] = A(16x1024) * V(1024x512) ----
  // V is staged 32 rows at a time into LDS by the Tensor Data Mover (TDM):
  // one contiguous 64 KB DMA per K-chunk, issued by wave 0, fenced with
  // s_wait_tensorcnt, published to the workgroup via barrier.
  float* Vstage = Ss;                                  // reuse dead 64 KB
  const unsigned ldsVoff = (unsigned)(size_t)(void*)Vstage; // LDS byte offset
  v8f accC[4];
#pragma unroll
  for (int p = 0; p < 4; ++p) accC[p] = (v8f){};

  const unsigned short* arow = &As[l16 * L_];

  for (int lblk = 0; lblk < L_; lblk += 32) {
    if (tid < 32) {                                    // wave 0 issues the DMA
      const unsigned long long ga =
          (unsigned long long)(size_t)(Vb + (size_t)lblk * H_);
      v4u g0;                                          // D# group 0 (ISA 8.3)
      g0[0] = 1u;                                      // count=1, user D#
      g0[1] = ldsVoff;                                 // lds_addr
      g0[2] = (unsigned)ga;                            // global_addr[31:0]
      g0[3] = (unsigned)((ga >> 32) & 0x01FFFFFFull)   // global_addr[56:32]
              | 0x80000000u;                           // type=2 ("image")
      v8i g1;                                          // D# group 1 (ISA 8.4)
      g1[0] = (2 << 16);                               // data_size=4B
      g1[1] = (int)((H_ & 0xFFFF) << 16);              // tensor_dim0 lo16
      g1[2] = (int)((H_ >> 16) | ((L_ & 0xFFFF) << 16)); // dim0 hi | dim1 lo
      g1[3] = (int)((L_ >> 16) | ((unsigned)H_ << 16));  // dim1 hi | tile_dim0
      g1[4] = 32;                                      // tile_dim1 = 32 rows
      g1[5] = H_;                                      // tensor_dim0_stride
      g1[6] = 0;
      g1[7] = 0;
      const v4i gz4 = {0, 0, 0, 0};                    // 2-D tile: groups 2/3 unused
      const v8i gz8 = {0, 0, 0, 0, 0, 0, 0, 0};
      // 6-arg form on this toolchain (clang-23 / therock-10.0 headers)
      __builtin_amdgcn_tensor_load_to_lds(g0, g1, gz4, gz4, gz8, 0);
      __builtin_amdgcn_s_wait_tensorcnt(0);
    }
    __syncthreads();                                   // tile visible to all waves

    // One A fragment (K = lblk..lblk+31) reused across all 4 h-passes.
    Frag16 a;
    a.q[0] = *(const uint4*)(arow + lblk + half * 16);
    a.q[1] = *(const uint4*)(arow + lblk + half * 16 + 8);

#pragma unroll
    for (int p = 0; p < 4; ++p) {
      const int hcol = p * 128 + wave * 16 + l16;
      Frag16 bm;
#pragma unroll
      for (int j = 0; j < 16; ++j)                     // B: col hcol, K rows of tile
        bm.u[j] = f2bf(Vstage[(half * 16 + j) * H_ + hcol]);
      accC[p] = wmma_bf16(a, bm, accC[p]);
    }
    __syncthreads();                                   // done reading before next DMA
  }

#pragma unroll
  for (int p = 0; p < 4; ++p) {
    const int hcol = p * 128 + wave * 16 + l16;
#pragma unroll
    for (int i = 0; i < 8; ++i)
      Cs[(i + 8 * half) * H_ + hcol] = f2bf(accC[p][i]);
  }
  __syncthreads();

  // ---- Phase D: out[16][C] = ctx(16x512) * W^T(512xC) + b; 7 waves cover C ----
  if (wave < 7) {
    const int c0   = wave * 16;
    const int ccol = c0 + l16;
    const unsigned short* crow = &Cs[l16 * H_];
    v8f acc = {};
    for (int h0 = 0; h0 < H_; h0 += 32) {
      Frag16 a, bm;
      a.q[0] = *(const uint4*)(crow + h0 + half * 16);
      a.q[1] = *(const uint4*)(crow + h0 + half * 16 + 8);
      if (ccol < C_) {
        const float* wp = Wg + (size_t)ccol * H_ + h0 + half * 16;
        cvt4(&bm.u[0],  ((const float4*)wp)[0]);
        cvt4(&bm.u[4],  ((const float4*)wp)[1]);
        cvt4(&bm.u[8],  ((const float4*)wp)[2]);
        cvt4(&bm.u[12], ((const float4*)wp)[3]);
      } else {
#pragma unroll
        for (int j = 0; j < 16; ++j) bm.u[j] = 0;
      }
      acc = wmma_bf16(a, bm, acc);   // wave-uniform branch above: EXEC all-1s
    }
    const float bias = (ccol < C_) ? bg[ccol] : 0.0f;
#pragma unroll
    for (int i = 0; i < 8; ++i) {
      const int t = t0 + i + 8 * half;
      if (t < T_ && ccol < C_)
        out[((size_t)b * T_ + t) * C_ + ccol] = acc[i] + bias;
    }
  }
}

extern "C" void kernel_launch(void* const* d_in, const int* in_sizes, int n_in,
                              void* d_out, int out_size, void* d_ws, size_t ws_size,
                              hipStream_t stream) {
  const float* Kg = (const float*)d_in[0];  // position_fmap [B,L,H]
  const float* Vg = (const float*)d_in[1];  // origin_fmap   [B,L,H]
  const float* Qg = (const float*)d_in[2];  // pos_emb       [T,H]
  const float* Wg = (const float*)d_in[3];  // W_gen         [C,H]
  const float* bg = (const float*)d_in[4];  // b_gen         [C]
  float* out = (float*)d_out;               // [B,T,C]

  dim3 grid(7 /* ceil(T/16) */, B_, 1);     // t-tiles fastest -> same-b tiles
  dim3 block(256, 1, 1);                    // co-resident, K/V[b] stays in L2
  attn_gen_kernel<<<grid, block, 0, stream>>>(Kg, Vg, Qg, Wg, bg, out);
}
